// Head_77369540870377
// MI455X (gfx1250) — matrix operand order repaired and checked by reading, not compiled, not run on hardware
//
#include <hip/hip_runtime.h>
#include <hip/hip_bf16.h>

typedef __attribute__((ext_vector_type(16))) __bf16 v16bf;
typedef __attribute__((ext_vector_type(8)))  __bf16 v8bf;
typedef __attribute__((ext_vector_type(8)))  float  v8f;

#define NE   1024   // n_embed
#define HS   128    // head size
#define TSEQ 2048   // T
#define BT   8192   // B*T
#define SPLITS 4    // key-split waves per query tile

// round-to-nearest-even fp32 -> bf16 (one-shot convert passes / epilogues only)
static __device__ inline __bf16 f2bf(float f) {
    union { float f; unsigned u; } in; in.f = f;
    unsigned r = in.u + 0x7FFFu + ((in.u >> 16) & 1u);
    union { unsigned short s; __bf16 b; } out; out.s = (unsigned short)(r >> 16);
    return out.b;
}

// 16-element WMMA operand from two contiguous 8-element spans (two b128 loads)
static __device__ inline v16bf load_frag(const __bf16* p0, const __bf16* p1) {
    union { v16bf f; v8bf h[2]; } u;
    u.h[0] = *(const v8bf*)p0;
    u.h[1] = *(const v8bf*)p1;
    return u.f;
}

// ---------------------------------------------------------------------------
// Prep 1: x fp32 [BT][NE] -> bf16 row-major.
// ---------------------------------------------------------------------------
__global__ __launch_bounds__(256) void cvt_x_kernel(
    const float* __restrict__ x, __bf16* __restrict__ xb)
{
    const size_t i = ((size_t)blockIdx.x * 256 + threadIdx.x) * 8;
    const float4 a = *(const float4*)(x + i);
    const float4 b = *(const float4*)(x + i + 4);
    v8bf o;
    o[0]=f2bf(a.x); o[1]=f2bf(a.y); o[2]=f2bf(a.z); o[3]=f2bf(a.w);
    o[4]=f2bf(b.x); o[5]=f2bf(b.y); o[6]=f2bf(b.z); o[7]=f2bf(b.w);
    *(v8bf*)(xb + i) = o;
}

// ---------------------------------------------------------------------------
// Prep 2: Wq/Wk/Wv fp32 [NE][HS] -> bf16 transposed WT[3][HS][NE].
// ---------------------------------------------------------------------------
__global__ __launch_bounds__(256) void cvt_w_kernel(
    const float* __restrict__ Wq, const float* __restrict__ Wk,
    const float* __restrict__ Wv, __bf16* __restrict__ wt)
{
    const int idx = (int)(blockIdx.x * 256 + threadIdx.x);  // 3*128*128 total
    const int k0  = (idx & 127) * 8;
    const int n   = (idx >> 7) & 127;
    const int mat = idx >> 14;
    const float* W = (mat == 0) ? Wq : ((mat == 1) ? Wk : Wv);
    v8bf o;
#pragma unroll
    for (int e = 0; e < 8; ++e) o[e] = f2bf(W[(size_t)(k0 + e) * HS + n]);
    *(v8bf*)(wt + ((size_t)mat * HS + n) * NE + k0) = o;
}

// ---------------------------------------------------------------------------
// Kernel 1: Q/K/V projection, all-bf16 operands.  One wave = 16-row x 128-col
// strip of one projection.  sched_barrier forces all 8 B-frags of a k-chunk
// into distinct registers before the WMMA group -> staged loadcnt waits.
// ---------------------------------------------------------------------------
__global__ __launch_bounds__(256) void qkv_kernel(
    const __bf16* __restrict__ xb, const __bf16* __restrict__ wt,
    __bf16* __restrict__ qo, __bf16* __restrict__ ko, __bf16* __restrict__ vo)
{
    const int wave  = (int)((blockIdx.x * blockDim.x + threadIdx.x) >> 5);
    const int lane  = (int)(threadIdx.x & 31);
    const int mat   = wave % 3;        // 0=Q, 1=K, 2=V
    const int mtile = wave / 3;        // 0..511
    const int m0    = mtile * 16;
    const int half  = lane >> 4;
    const int n     = lane & 15;

    const v8f vzero = {0.f,0.f,0.f,0.f,0.f,0.f,0.f,0.f};
    v8f acc[8];
#pragma unroll
    for (int t = 0; t < 8; ++t) acc[t] = vzero;

    const __bf16* xrow = xb + (size_t)(m0 + n) * NE;
    const __bf16* wbase = wt + ((size_t)mat * HS + n) * NE;

    for (int kc = 0; kc < NE; kc += 32) {
        const v16bf afrag = load_frag(xrow + kc + half * 8,
                                      xrow + kc + 16 + half * 8);
        v16bf bf[8];
#pragma unroll
        for (int nt = 0; nt < 8; ++nt) {
            const __bf16* wp = wbase + (size_t)nt * 16 * NE + kc;
            bf[nt] = load_frag(wp + half * 8, wp + 16 + half * 8);
        }
        __builtin_amdgcn_sched_barrier(0);   // keep all loads ahead of WMMAs
#pragma unroll
        for (int nt = 0; nt < 8; ++nt)
            acc[nt] = __builtin_amdgcn_wmma_f32_16x16x32_bf16(
                false, afrag, false, bf[nt], (short)0, acc[nt], false, false);
    }

    // D layout: VGPR r -> row M = r + 8*half, col N = lane%16
    if (mat == 2) {
#pragma unroll
        for (int nt = 0; nt < 8; ++nt)
#pragma unroll
            for (int r = 0; r < 8; ++r)
                vo[(size_t)(nt * 16 + n) * BT + m0 + r + 8 * half] = f2bf(acc[nt][r]);
    } else {
        __bf16* dst = (mat == 0) ? qo : ko;
#pragma unroll
        for (int nt = 0; nt < 8; ++nt)
#pragma unroll
            for (int r = 0; r < 8; ++r)
                dst[(size_t)(m0 + r + 8 * half) * HS + nt * 16 + n] = f2bf(acc[nt][r]);
    }
}

// ---------------------------------------------------------------------------
// Kernel 2: causal flash attention, 4-way key split per query tile.
// ---------------------------------------------------------------------------
__global__ __launch_bounds__(256) void attn_kernel(
    const __bf16* __restrict__ q, const __bf16* __restrict__ k,
    const __bf16* __restrict__ vT, float* __restrict__ out)
{
    __shared__ __align__(16) float  sO[2][SPLITS][16][HS];   // 64 KB partials
    __shared__ float                sM[2][SPLITS][16];
    __shared__ float                sL[2][SPLITS][16];
    __shared__ __align__(16) __bf16 sP[8][16 * 32];          // per-wave P tile

    const int wib   = (int)(threadIdx.x >> 5);
    const int lane  = (int)(threadIdx.x & 31);
    const int half  = lane >> 4;
    const int n     = lane & 15;
    const int tt    = wib >> 2;                 // tile slot in block (0/1)
    const int split = wib & 3;

    const int gt = (int)blockIdx.x * 2 + tt;    // global query tile 0..511
    const int bb = gt >> 7;
    const int q0 = (gt & 127) * 16;
    const size_t base = (size_t)bb * TSEQ;

    // Q fragments (A layout, row M = lane%16)
    v16bf aq[4];
    {
        const __bf16* qrow = q + (base + q0 + n) * (size_t)HS;
#pragma unroll
        for (int t = 0; t < 4; ++t)
            aq[t] = load_frag(qrow + t * 32 + half * 8,
                              qrow + t * 32 + 16 + half * 8);
    }

    const v8f vzero = {0.f,0.f,0.f,0.f,0.f,0.f,0.f,0.f};
    float mrow[8], lrow[8];
    v8f o[8];
#pragma unroll
    for (int r = 0; r < 8; ++r) { mrow[r] = -1e30f; lrow[r] = 0.f; }
#pragma unroll
    for (int t = 0; t < 8; ++t) o[t] = vzero;

    const float scale = 0.03125f;               // 1024^-0.5
    const float LOG2E = 1.4426950408889634f;

    const int kend = q0 + 16;
    for (int k0 = split * 32; k0 < kend; k0 += SPLITS * 32) {
        // prefetch this split's next K chunk (global_prefetch_b8)
        if (k0 + SPLITS * 32 < kend)
            __builtin_prefetch(k + (base + k0 + SPLITS * 32 + n) * (size_t)HS, 0, 1);

        // ---- batch all 8 K-fragment loads, then 8 S WMMAs ----------------
        const __bf16* krow0 = k + (base + k0 + n) * (size_t)HS + half * 8;
        const __bf16* krow1 = k + (base + k0 + 16 + n) * (size_t)HS + half * 8;
        v16bf kb0[4], kb1[4];
#pragma unroll
        for (int t = 0; t < 4; ++t) {
            kb0[t] = load_frag(krow0 + t * 32, krow0 + t * 32 + 16);
            kb1[t] = load_frag(krow1 + t * 32, krow1 + t * 32 + 16);
        }
        __builtin_amdgcn_sched_barrier(0);   // loads stay ahead of WMMAs
        v8f s0 = vzero, s1 = vzero;
#pragma unroll
        for (int t = 0; t < 4; ++t) {
            s0 = __builtin_amdgcn_wmma_f32_16x16x32_bf16(false, aq[t], false, kb0[t], (short)0, s0, false, false);
            s1 = __builtin_amdgcn_wmma_f32_16x16x32_bf16(false, aq[t], false, kb1[t], (short)0, s1, false, false);
        }

        // ---- issue V-fragment loads now; latency hides under softmax -----
        v16bf vb[8];
#pragma unroll
        for (int nt = 0; nt < 8; ++nt) {
            const __bf16* vp = vT + (size_t)(nt * 16 + n) * BT + base + k0 + half * 8;
            vb[nt] = load_frag(vp, vp + 16);
        }
        __builtin_amdgcn_sched_barrier(0);   // don't sink V loads below softmax

        // ---- scale, causal mask, online softmax --------------------------
        float alpha[8];
#pragma unroll
        for (int r = 0; r < 8; ++r) {
            const int qg = q0 + r + 8 * half;
            float v0 = s0[r] * scale; if (k0 + n > qg)      v0 = -1e30f;
            float v1 = s1[r] * scale; if (k0 + 16 + n > qg) v1 = -1e30f;
            float mx = fmaxf(v0, v1);
#pragma unroll
            for (int off = 1; off < 16; off <<= 1)
                mx = fmaxf(mx, __shfl_xor(mx, off, 32));
            const float mnew = fmaxf(mrow[r], mx);
            const float a  = exp2f((mrow[r] - mnew) * LOG2E);
            const float p0 = exp2f((v0 - mnew) * LOG2E);
            const float p1 = exp2f((v1 - mnew) * LOG2E);
            float rs = p0 + p1;
#pragma unroll
            for (int off = 1; off < 16; off <<= 1)
                rs += __shfl_xor(rs, off, 32);
            lrow[r] = lrow[r] * a + rs;
            mrow[r] = mnew;
            alpha[r] = a;
            const int m = r + 8 * half;
            sP[wib][m * 32 + n]      = f2bf(p0);
            sP[wib][m * 32 + 16 + n] = f2bf(p1);
        }

#pragma unroll
        for (int nt = 0; nt < 8; ++nt)
#pragma unroll
            for (int r = 0; r < 8; ++r) o[nt][r] *= alpha[r];

        // ---- P back in A layout from LDS (in-order DS per wave) ----------
        const __bf16* pp = &sP[wib][n * 32];
        const v16bf ap = load_frag(pp + half * 8, pp + 16 + half * 8);

        // ---- O += P*V ----------------------------------------------------
#pragma unroll
        for (int nt = 0; nt < 8; ++nt)
            o[nt] = __builtin_amdgcn_wmma_f32_16x16x32_bf16(
                false, ap, false, vb[nt], (short)0, o[nt], false, false);
    }

    // ---- publish partials ------------------------------------------------
#pragma unroll
    for (int nt = 0; nt < 8; ++nt)
#pragma unroll
        for (int r = 0; r < 8; ++r)
            sO[tt][split][r + 8 * half][nt * 16 + n] = o[nt][r];
    if (n == 0) {
#pragma unroll
        for (int r = 0; r < 8; ++r) {
            sM[tt][split][r + 8 * half] = mrow[r];
            sL[tt][split][r + 8 * half] = lrow[r];
        }
    }
    __syncthreads();

    // ---- merge 4 partials, normalize, write out [B,T,H] ------------------
#pragma unroll 4
    for (int it = 0; it < 16; ++it) {
        const int idx = it * 256 + (int)threadIdx.x;   // 0..4095, h fastest
        const int tt2 = idx >> 11;
        const int m   = (idx >> 7) & 15;
        const int h   = idx & 127;
        float mstar = sM[tt2][0][m];
#pragma unroll
        for (int i = 1; i < SPLITS; ++i) mstar = fmaxf(mstar, sM[tt2][i][m]);
        float lsum = 0.f, acc = 0.f;
#pragma unroll
        for (int i = 0; i < SPLITS; ++i) {
            const float w = exp2f((sM[tt2][i][m] - mstar) * LOG2E);
            lsum += w * sL[tt2][i][m];
            acc  += w * sO[tt2][i][m][h];
        }
        const int gt2 = (int)blockIdx.x * 2 + tt2;
        out[((size_t)(gt2 >> 7) * TSEQ + (gt2 & 127) * 16 + m) * (size_t)HS + h]
            = acc / lsum;
    }
}

// ---------------------------------------------------------------------------
extern "C" void kernel_launch(void* const* d_in, const int* in_sizes, int n_in,
                              void* d_out, int out_size, void* d_ws, size_t ws_size,
                              hipStream_t stream) {
    const float* x  = (const float*)d_in[0];
    const float* Wq = (const float*)d_in[1];
    const float* Wk = (const float*)d_in[2];
    const float* Wv = (const float*)d_in[3];

    __bf16* qws = (__bf16*)d_ws;                   // [BT][HS]
    __bf16* kws = qws + (size_t)BT * HS;           // [BT][HS]
    __bf16* vws = kws + (size_t)BT * HS;           // [HS][BT] transposed
    __bf16* xbf = vws + (size_t)BT * HS;           // [BT][NE]
    __bf16* wtb = xbf + (size_t)BT * NE;           // [3][HS][NE] transposed

    cvt_x_kernel<<<(BT * NE) / (256 * 8), 256, 0, stream>>>(x, xbf);
    cvt_w_kernel<<<(3 * HS * HS) / 256, 256, 0, stream>>>(Wq, Wk, Wv, wtb);
    qkv_kernel<<<192, 256, 0, stream>>>(xbf, wtb, qws, kws, vws);
    attn_kernel<<<256, 256, 0, stream>>>(qws, kws, vws, (float*)d_out);

    (void)in_sizes; (void)n_in; (void)out_size; (void)ws_size;
}
